// HMPNN_sum_2Layer_53798760349845
// MI455X (gfx1250) — compile-verified, matrix operand order checked by reading, not run on hardware
//
#include <hip/hip_runtime.h>
#include <hip/hip_bf16.h>

// ---------------------------------------------------------------------------
// HMPNN (hetero NNConv x2 + 1-channel output layer) for MI455X / gfx1250.
//
// Per-edge matmul rewritten as GEMM:  msg = Z @ Wt,
//   Z[e] = [ vec(x_src[e] (x) eattr[e])  (256) ,  x_src[e] (32) ]   K = 288
//   Wt[k= i*8+a][o] = Wm[a][i*32+o],  Wt[256+i][o] = bm[i*32+o]
// Executed as 16-edge tiles with v_wmma_f32_16x16x32_f16:
//   9 K-steps x 2 N-tiles = 18 WMMA per wave-tile, f32 accumulation.
// Wt is pre-swizzled into the per-lane WMMA B-fragment layout so each lane
// loads its v16h as a contiguous 32B chunk (coalesced, L2-resident).
// ---------------------------------------------------------------------------

typedef __attribute__((ext_vector_type(16))) _Float16 v16h;
typedef __attribute__((ext_vector_type(8)))  float    v8f;

#define F_DIM 32
#define EA_DIM 8
#define K_DIM 288            // 256 (x (x) e) + 32 (bm/x)
#define BPACK_HALFS (9 * 2 * 32 * 16)   // steps * ntiles * lanes * halfs = 9216

// select xr[i0] (lanes 0-15) vs xr[i1] (lanes 16-31); i0,i1 compile-time under
// full unroll -> v_cndmask, no dynamic VGPR indexing / scratch.
__device__ __forceinline__ _Float16 hsel(const _Float16* x, int i0, int i1, int hi) {
  return hi ? x[i1] : x[i0];
}

// --- Pre-swizzle Wt (from Wm[EA,F*F], bm[F*F]) into WMMA B-fragment order ---
// B fragment (16x16x32 f16): lane L holds column n = L%16; half j maps to
// K = kbase + (j>>3)*16 + 8*(L>=16) + (j&7).  Store as
// Bpack[((s*2 + t)*32 + L)*16 + j]  (t = n/16, s = k/32).
__global__ void build_bpack_k(const float* __restrict__ Wm,
                              const float* __restrict__ bm,
                              _Float16* __restrict__ Bpack) {
  int idx = blockIdx.x * blockDim.x + threadIdx.x;   // over 288*32
  if (idx >= K_DIM * F_DIM) return;
  int k = idx >> 5;          // 0..287
  int n = idx & 31;          // 0..31
  float v;
  if (k < 256) {
    int i = k >> 3, a = k & 7;
    v = Wm[a * (F_DIM * F_DIM) + i * F_DIM + n];
  } else {
    v = bm[(k - 256) * F_DIM + n];
  }
  int s   = k >> 5;
  int rem = k & 31;
  int g   = rem >> 4;        // -> j>>3
  int hi  = (rem >> 3) & 1;  // -> lane>=16
  int r   = rem & 7;         // -> j&7
  int j   = g * 8 + r;
  int L   = hi * 16 + (n & 15);
  int t   = n >> 4;
  Bpack[((s * 2 + t) * 32 + L) * 16 + j] = (_Float16)v;
}

// --- Root term: acc[n][o] = b[o] + sum_i x[n][i] * Wr[i][o] ----------------
__global__ void root_mm32_k(const float* __restrict__ x,
                            const float* __restrict__ Wr,
                            const float* __restrict__ b,
                            float* __restrict__ acc, int N) {
  int idx = blockIdx.x * blockDim.x + threadIdx.x;
  if (idx >= N * F_DIM) return;
  int n = idx >> 5, o = idx & 31;
  const float* xr = x + (long long)n * F_DIM;
  float s = b[o];
#pragma unroll
  for (int i = 0; i < F_DIM; ++i) s += xr[i] * Wr[i * F_DIM + o];
  acc[idx] = s;
}

// --- WMMA edge kernel: one wave per 16-edge tile ---------------------------
__global__ __launch_bounds__(256) void edge_nnconv_wmma_k(
    const float* __restrict__ xsrc, const int* __restrict__ src,
    const int* __restrict__ dst, const float* __restrict__ eattr,
    const _Float16* __restrict__ Bpack, float* __restrict__ acc, int E) {
  const int lane  = threadIdx.x & 31;
  const int tile  = blockIdx.x * 8 + (threadIdx.x >> 5);
  const int ntile = (E + 15) >> 4;
  if (tile >= ntile) return;                 // wave-uniform exit, EXEC stays full

  const int  m  = lane & 15;                 // edge row inside tile
  const int  hi = lane >> 4;                 // K-half select
  const int  e  = tile * 16 + m;
  const bool ok = (e < E);
  const int  se = ok ? src[e] : 0;
  const int  de = ok ? dst[e] : 0;

  // lane-private copies (lanes 16-31 duplicate rows 0-15; L1/L2 absorbs it)
  _Float16 xr[F_DIM];
  const float* xp = xsrc + (long long)se * F_DIM;
#pragma unroll
  for (int i = 0; i < F_DIM; ++i) xr[i] = ok ? (_Float16)xp[i] : (_Float16)0.f;
  _Float16 eh[EA_DIM];
  const float* ep = eattr + (long long)e * EA_DIM;
#pragma unroll
  for (int a = 0; a < EA_DIM; ++a) eh[a] = ok ? (_Float16)ep[a] : (_Float16)0.f;

  v8f c0 = {};   // out channels 0..15
  v8f c1 = {};   // out channels 16..31
#pragma unroll
  for (int s = 0; s < 9; ++s) {
    // Build A fragment: half j <-> K = s*32 + (j>>3)*16 + 8*hi + (j&7).
    // K<256: Z = x[K>>3] * e[K&7];  K>=256: Z = x[K-256].
    v16h a;
#pragma unroll
    for (int j = 0; j < 16; ++j) {
      if (s < 8) {
        int i0 = s * 4 + (j >> 3) * 2;                // x index = i0 + hi
        a[j] = hsel(xr, i0, i0 + 1, hi) * eh[j & 7];
      } else {
        int i0 = (j >> 3) * 16 + (j & 7);             // x index = i0 + 8*hi
        a[j] = hsel(xr, i0, i0 + 8, hi);
      }
    }
    const v16h* bp = (const v16h*)Bpack + (s * 2) * 32 + lane;
    v16h b0 = bp[0];        // contiguous 32B per lane, 1KB per wave
    v16h b1 = bp[32];
    c0 = __builtin_amdgcn_wmma_f32_16x16x32_f16(false, a, false, b0,
                                                (short)0, c0, false, false);
    c1 = __builtin_amdgcn_wmma_f32_16x16x32_f16(false, a, false, b1,
                                                (short)0, c1, false, false);
  }

  // D layout: lane holds col n = lane&15; VGPR v holds row M = v + 8*hi.
  // Padded edge rows produced all-zero A -> zero msg -> harmless +0.0 adds.
#pragma unroll
  for (int v = 0; v < 8; ++v) {
    int mrow = v + 8 * hi;
    int d    = __shfl(de, mrow, 32);
    atomicAdd(acc + (long long)d * F_DIM + m,      c0[v]);
    atomicAdd(acc + (long long)d * F_DIM + 16 + m, c1[v]);
  }
}

// --- elementwise sigmoid ---------------------------------------------------
__global__ void sigmoid_inplace_k(float* __restrict__ p, long long n) {
  long long i = blockIdx.x * (long long)blockDim.x + threadIdx.x;
  if (i < n) p[i] = 1.f / (1.f + __expf(-p[i]));
}
__global__ void sigmoid_to_k(const float* __restrict__ in, float* __restrict__ out,
                             int n) {
  int i = blockIdx.x * blockDim.x + threadIdx.x;
  if (i < n) out[i] = 1.f / (1.f + __expf(-in[i]));
}

// --- output layer root: acc[n] = b2 + h[n]·Wr2 -----------------------------
__global__ void out_root_k(const float* __restrict__ h, const float* __restrict__ Wr2,
                           const float* __restrict__ b2, float* __restrict__ acc,
                           int N) {
  int n = blockIdx.x * blockDim.x + threadIdx.x;
  if (n >= N) return;
  const float* hr = h + (long long)n * F_DIM;
  float s = b2[0];
#pragma unroll
  for (int i = 0; i < F_DIM; ++i) s += hr[i] * Wr2[i];
  acc[n] = s;
}

// --- output layer edges (f_out = 1): msg = h·bm2 + Σ_a e_a (h·Wm2_a) -------
__global__ void out_edge_k(const float* __restrict__ h, const int* __restrict__ src,
                           const int* __restrict__ dst, const float* __restrict__ eattr,
                           const float* __restrict__ Wm2, const float* __restrict__ bm2,
                           float* __restrict__ acc, int E) {
  int e = blockIdx.x * blockDim.x + threadIdx.x;
  if (e >= E) return;
  const float* hr = h + (long long)src[e] * F_DIM;
  const float* ep = eattr + (long long)e * EA_DIM;
  float da[EA_DIM] = {};
  float db = 0.f;
#pragma unroll
  for (int i = 0; i < F_DIM; ++i) {
    float hv = hr[i];
    db += hv * bm2[i];
#pragma unroll
    for (int a = 0; a < EA_DIM; ++a) da[a] += hv * Wm2[a * F_DIM + i];
  }
  float msg = db;
#pragma unroll
  for (int a = 0; a < EA_DIM; ++a) msg += ep[a] * da[a];
  atomicAdd(acc + dst[e], msg);
}

// ---------------------------------------------------------------------------
extern "C" void kernel_launch(void* const* d_in, const int* in_sizes, int n_in,
                              void* d_out, int out_size, void* d_ws, size_t ws_size,
                              hipStream_t stream) {
  const float* x_indivi = (const float*)d_in[0];
  const float* x_org    = (const float*)d_in[1];
  const int*   esrc_oi  = (const int*)d_in[2];
  const int*   edst_oi  = (const int*)d_in[3];
  const float* eatt_oi  = (const float*)d_in[4];
  const int*   esrc_io  = (const int*)d_in[5];
  const int*   edst_io  = (const int*)d_in[6];
  const float* eatt_io  = (const float*)d_in[7];
  const float* Wm_oi = (const float*)d_in[8];
  const float* bm_oi = (const float*)d_in[9];
  const float* Wr_oi = (const float*)d_in[10];
  const float* b_oi  = (const float*)d_in[11];
  const float* Wm_io = (const float*)d_in[12];
  const float* bm_io = (const float*)d_in[13];
  const float* Wr_io = (const float*)d_in[14];
  const float* b_io  = (const float*)d_in[15];
  const float* Wm2   = (const float*)d_in[16];
  const float* bm2   = (const float*)d_in[17];
  const float* Wr2   = (const float*)d_in[18];
  const float* b2    = (const float*)d_in[19];

  const int NI   = in_sizes[0] / F_DIM;
  const int NO   = in_sizes[1] / F_DIM;
  const int E_oi = in_sizes[2];
  const int E_io = in_sizes[5];

  // workspace carve (256B aligned)
  size_t off = 0;
  auto carve = [&](size_t bytes) -> void* {
    void* p = (char*)d_ws + off;
    off += (bytes + 255) & ~(size_t)255;
    return p;
  };
  float*    hI     = (float*)carve((size_t)NI * F_DIM * sizeof(float));
  float*    hO     = (float*)carve((size_t)NO * F_DIM * sizeof(float));
  float*    outacc = (float*)carve((size_t)NI * sizeof(float));
  _Float16* bpOI   = (_Float16*)carve((size_t)BPACK_HALFS * sizeof(_Float16));
  _Float16* bpIO   = (_Float16*)carve((size_t)BPACK_HALFS * sizeof(_Float16));
  (void)ws_size; (void)n_in; (void)out_size;

  const int BT = 256;
  auto cdiv = [](long long a, long long b) { return (int)((a + b - 1) / b); };

  // 1) pre-swizzled WMMA B fragments for both relations
  build_bpack_k<<<cdiv(K_DIM * F_DIM, BT), BT, 0, stream>>>(Wm_oi, bm_oi, bpOI);
  build_bpack_k<<<cdiv(K_DIM * F_DIM, BT), BT, 0, stream>>>(Wm_io, bm_io, bpIO);

  // 2) root terms into accumulators
  root_mm32_k<<<cdiv((long long)NI * F_DIM, BT), BT, 0, stream>>>(x_indivi, Wr_oi, b_oi, hI, NI);
  root_mm32_k<<<cdiv((long long)NO * F_DIM, BT), BT, 0, stream>>>(x_org,    Wr_io, b_io, hO, NO);

  // 3) per-edge WMMA message passing + atomic scatter-add
  int blk_oi = cdiv((E_oi + 15) >> 4, 8);   // 8 waves (tiles) per 256-thread block
  int blk_io = cdiv((E_io + 15) >> 4, 8);
  edge_nnconv_wmma_k<<<blk_oi, BT, 0, stream>>>(x_org,    esrc_oi, edst_oi, eatt_oi, bpOI, hI, E_oi);
  edge_nnconv_wmma_k<<<blk_io, BT, 0, stream>>>(x_indivi, esrc_io, edst_io, eatt_io, bpIO, hO, E_io);

  // 4) layer-1 sigmoid
  sigmoid_inplace_k<<<cdiv((long long)NI * F_DIM, BT), BT, 0, stream>>>(hI, (long long)NI * F_DIM);
  sigmoid_inplace_k<<<cdiv((long long)NO * F_DIM, BT), BT, 0, stream>>>(hO, (long long)NO * F_DIM);

  // 5) output layer (f_out = 1): root + edge scatter + sigmoid into d_out
  out_root_k<<<cdiv(NI, BT), BT, 0, stream>>>(hI, Wr2, b2, outacc, NI);
  out_edge_k<<<cdiv(E_oi, BT), BT, 0, stream>>>(hO, esrc_oi, edst_oi, eatt_oi, Wm2, bm2, outacc, E_oi);
  sigmoid_to_k<<<cdiv(NI, BT), BT, 0, stream>>>(outacc, (float*)d_out, NI);
}
